// BoundaryDecoderAttention_3547642987252
// MI455X (gfx1250) — compile-verified, matrix operand order checked by reading, not run on hardware
//
#include <hip/hip_runtime.h>
#include <hip/hip_bf16.h>

typedef __attribute__((ext_vector_type(16))) _Float16 v16h;
typedef __attribute__((ext_vector_type(8)))  _Float16 v8h;
typedef __attribute__((ext_vector_type(8)))  float    v8f;

#define B_   64
#define T_   4096
#define D_   256
#define TILE_T 64      // T rows per block in main kernel
#define KSTEPS 8       // 256 / 32

__device__ __forceinline__ float fast_tanh(float x) {
#if __has_builtin(__builtin_amdgcn_tanhf)
    return __builtin_amdgcn_tanhf(x);          // v_tanh_f32 (gfx1250 trans op)
#elif __has_builtin(__builtin_amdgcn_tanh_f32)
    return __builtin_amdgcn_tanh_f32(x);
#else
    return tanhf(x);
#endif
}

// ---------------------------------------------------------------------------
// Kernel 1: pack V_w (f32, [o,i] row-major) into f16 WMMA B-fragments.
// B[k][n] = V_w[n][k].  Fragment layout (per 32x16 KxN tile, per lane, 16 halves):
//   lane<16 : n = nt*16+lane,    halves 0..7 -> K=ks*32+0..7,   8..15 -> K=ks*32+16..23
//   lane>=16: n = nt*16+lane-16, halves 0..7 -> K=ks*32+8..15,  8..15 -> K=ks*32+24..31
// Flat: pB[((nt*8+ks)*32 + lane)*16 + h]
// ---------------------------------------------------------------------------
__global__ void pack_b_kernel(const float* __restrict__ Vw, _Float16* __restrict__ pB) {
    int gid = blockIdx.x * 256 + threadIdx.x;      // 64 blocks * 256 threads
    int f   = gid * 4;                              // 4 halves per thread, 65536 total
    int h4   = f & 15;                              // 0,4,8,12
    int lane = (f >> 4) & 31;
    int ks   = (f >> 9) & 7;
    int nt   = f >> 12;
    int n    = nt * 16 + (lane & 15);
    int koff = ((lane >> 4) & 1) * 8 + ((h4 >> 3) & 1) * 16;
    int k0   = ks * 32 + koff + (h4 & 4);
    const float4 src = *(const float4*)(Vw + (size_t)n * D_ + k0);
    _Float16 out[4] = { (_Float16)src.x, (_Float16)src.y, (_Float16)src.z, (_Float16)src.w };
    *(uint2*)(pB + f) = *(uint2*)out;
}

// ---------------------------------------------------------------------------
// Kernel 2: Fk_prime[b,o] = sum_i h_tm1[b,i]*Wa_w[o,i] + Wa_b[o]
//           + zero-init z_num accumulator (exactly B*D threads)
// ---------------------------------------------------------------------------
__global__ void fkp_kernel(const float* __restrict__ h_tm1, const float* __restrict__ Wa_w,
                           const float* __restrict__ Wa_b, float* __restrict__ fkp,
                           float* __restrict__ z_num) {
    __shared__ float sh[D_];
    int b = blockIdx.x, o = threadIdx.x;
    z_num[b * D_ + o] = 0.0f;
    sh[o] = h_tm1[b * D_ + o];
    __syncthreads();
    float s = Wa_b[o];
    const float* wr = Wa_w + (size_t)o * D_;
    #pragma unroll 8
    for (int i = 0; i < D_; ++i) s += sh[i] * wr[i];
    fkp[b * D_ + o] = s;
}

// ---------------------------------------------------------------------------
// Kernel 3: fused GEMM(WMMA f16) + tanh + dot(v) + clip/mask -> beta_raw[b,t],
//           plus shifted-exp weighted tile sum  z_num[b,:] += sum_t e'_t * H_r[b,t,:]
//           with e'_t = exp(beta_t - 15)*m_t  (beta clipped to <=15, so safe).
// grid = B * (T/TILE_T) = 4096 blocks, 256 threads (8 waves)
// ---------------------------------------------------------------------------
__global__ void __launch_bounds__(256)
main_kernel(const float* __restrict__ H_r, const int* __restrict__ mask,
            const float* __restrict__ fkp, const _Float16* __restrict__ pB,
            const float* __restrict__ Vb, const float* __restrict__ vvec,
            const float* __restrict__ cc, float* __restrict__ beta_raw,
            float* __restrict__ z_num) {
    __shared__ _Float16 a_lds[4 * KSTEPS * 32 * 16];   // 32 KB, A fragments
    __shared__ float    s_row[TILE_T];
    __shared__ float    s_e[TILE_T];

    const int tid  = threadIdx.x;
    const int wave = tid >> 5;
    const int lane = tid & 31;
    const int b    = blockIdx.x >> 6;            // T/TILE_T = 64 tiles
    const int t0   = (blockIdx.x & 63) * TILE_T;

    if (tid < TILE_T) s_row[tid] = 0.0f;

    // ---- stage A tile: 64 rows x 256 cols f32 -> f16 fragments in LDS ----
    #pragma unroll
    for (int i = 0; i < 8; ++i) {
        int u   = tid + (i << 8);                // 2048 units: (row, 8-wide K chunk)
        int row = u >> 5;
        int ch  = u & 31;
        const float4* src = (const float4*)(H_r + ((size_t)(b * T_ + t0 + row)) * D_ + ch * 8);
        float4 f0 = src[0], f1 = src[1];
        v8h hv;
        hv[0] = (_Float16)f0.x; hv[1] = (_Float16)f0.y; hv[2] = (_Float16)f0.z; hv[3] = (_Float16)f0.w;
        hv[4] = (_Float16)f1.x; hv[5] = (_Float16)f1.y; hv[6] = (_Float16)f1.z; hv[7] = (_Float16)f1.w;
        int mt     = row >> 4;
        int ks     = ch >> 2;
        int lane_d = (row & 15) + ((ch & 1) << 4);
        int hb     = ((ch >> 1) & 1) << 3;
        *(v8h*)&a_lds[(((mt * KSTEPS + ks) * 32 + lane_d) << 4) + hb] = hv;
    }
    __syncthreads();

    // ---- WMMA: each wave owns n-tiles 2w and 2w+1, all 4 m-tiles ----
    const int nt0 = 2 * wave, nt1 = 2 * wave + 1;
    v8f acc[4][2];
    #pragma unroll
    for (int m = 0; m < 4; ++m) { acc[m][0] = (v8f)0.0f; acc[m][1] = (v8f)0.0f; }

    #pragma unroll
    for (int ks = 0; ks < KSTEPS; ++ks) {
        const v16h bf0 = *(const v16h*)(pB + (((nt0 * KSTEPS + ks) * 32 + lane) << 4));
        const v16h bf1 = *(const v16h*)(pB + (((nt1 * KSTEPS + ks) * 32 + lane) << 4));
        #pragma unroll
        for (int mt = 0; mt < 4; ++mt) {
            const v16h af = *(const v16h*)&a_lds[((mt * KSTEPS + ks) * 32 + lane) << 4];
            acc[mt][0] = __builtin_amdgcn_wmma_f32_16x16x32_f16(false, af, false, bf0,
                                                                (short)0, acc[mt][0], false, false);
            acc[mt][1] = __builtin_amdgcn_wmma_f32_16x16x32_f16(false, af, false, bf1,
                                                                (short)0, acc[mt][1], false, false);
        }
    }

    // ---- epilogue: tanh(acc + Vb + fkp) * v[n], reduce over n ----
    const int lh     = lane & 15;
    const int half16 = lane >> 4;                 // 0: rows r, 1: rows 8+r
    const int n0 = nt0 * 16 + lh, n1 = nt1 * 16 + lh;
    const float bias0 = Vb[n0] + fkp[b * D_ + n0];
    const float bias1 = Vb[n1] + fkp[b * D_ + n1];
    const float vv0 = vvec[n0], vv1 = vvec[n1];

    #pragma unroll
    for (int mt = 0; mt < 4; ++mt) {
        #pragma unroll
        for (int r = 0; r < 8; ++r) {
            float p = fast_tanh(acc[mt][0][r] + bias0) * vv0
                    + fast_tanh(acc[mt][1][r] + bias1) * vv1;
            p += __shfl_xor(p, 1);
            p += __shfl_xor(p, 2);
            p += __shfl_xor(p, 4);
            p += __shfl_xor(p, 8);                // full sum within each 16-lane half
            if (lh == 0) atomicAdd(&s_row[mt * 16 + half16 * 8 + r], p);
        }
    }
    __syncthreads();

    if (tid < TILE_T) {
        float raw = s_row[tid] + cc[0];
        raw = fminf(fmaxf(raw, -15.0f), 15.0f);
        const float m = (float)mask[b * T_ + t0 + tid];
        raw *= m;                                 // mask before exp (faithful)
        beta_raw[b * T_ + t0 + tid] = raw;
        s_e[tid] = __expf(raw - 15.0f) * m;       // shifted exp weight (<= 1)
    }
    __syncthreads();

    // ---- weighted tile sum: z_num[b,i] += sum_row e'_row * H_r[b,t0+row,i] ----
    // (tile is cache-hot: this block just streamed it)
    const float* Hb = H_r + ((size_t)(b * T_ + t0)) * D_;
    float zacc = 0.0f;
    #pragma unroll 8
    for (int row = 0; row < TILE_T; ++row)
        zacc = fmaf(s_e[row], Hb[(size_t)row * D_ + tid], zacc);
    unsafeAtomicAdd(&z_num[b * D_ + tid], zacc);
}

// ---------------------------------------------------------------------------
// Kernel 4: per-batch exact softmax over T (true max) -> beta output,
//           plus rescale of shifted z accumulator:
//           z = z_num * exp(15 - mx) / (sum + 1e-6)
// grid = B blocks, 256 threads; each thread handles 16 strided t's
// ---------------------------------------------------------------------------
__global__ void finalize_kernel(const float* __restrict__ beta_raw, const int* __restrict__ mask,
                                const float* __restrict__ z_num, float* __restrict__ d_out) {
    __shared__ float red[256];
    const int b = blockIdx.x, tid = threadIdx.x;

    float vals[16];
    float mx = -3.4e38f;
    #pragma unroll
    for (int j = 0; j < 16; ++j) {
        float x = beta_raw[b * T_ + tid + j * 256];
        vals[j] = x;
        mx = fmaxf(mx, x);
    }
    red[tid] = mx; __syncthreads();
    for (int s = 128; s > 0; s >>= 1) { if (tid < s) red[tid] = fmaxf(red[tid], red[tid + s]); __syncthreads(); }
    const float smax = red[0];
    __syncthreads();

    float sum = 0.0f;
    #pragma unroll
    for (int j = 0; j < 16; ++j) {
        float e = __expf(vals[j] - smax) * (float)mask[b * T_ + tid + j * 256];
        vals[j] = e;
        sum += e;
    }
    red[tid] = sum; __syncthreads();
    for (int s = 128; s > 0; s >>= 1) { if (tid < s) red[tid] += red[tid + s]; __syncthreads(); }
    const float inv = 1.0f / (red[0] + 1e-6f);

    #pragma unroll
    for (int j = 0; j < 16; ++j)
        d_out[B_ * D_ + b * T_ + tid + j * 256] = vals[j] * inv;

    // z finalize: Σ_t exp(β_t - mx)·H_t = exp(15 - mx)·z_num  (β ≤ 15 by clip)
    const float zscale = __expf(15.0f - smax) * inv;
    d_out[b * D_ + tid] = z_num[b * D_ + tid] * zscale;
}

// ---------------------------------------------------------------------------
extern "C" void kernel_launch(void* const* d_in, const int* in_sizes, int n_in,
                              void* d_out, int out_size, void* d_ws, size_t ws_size,
                              hipStream_t stream) {
    const float* H_r   = (const float*)d_in[0];
    const int*   mask  = (const int*)  d_in[1];
    const float* h_tm1 = (const float*)d_in[2];
    const float* V_w   = (const float*)d_in[3];
    const float* V_b   = (const float*)d_in[4];
    const float* Wa_w  = (const float*)d_in[5];
    const float* Wa_b  = (const float*)d_in[6];
    const float* vvec  = (const float*)d_in[7];
    const float* c     = (const float*)d_in[8];
    float* out = (float*)d_out;

    // workspace layout
    float*     fkp      = (float*)d_ws;                                       // 64KB
    _Float16*  pB       = (_Float16*)((char*)d_ws + 65536);                   // 128KB
    float*     beta_raw = (float*)((char*)d_ws + 65536 + 131072);             // 1MB
    float*     z_num    = (float*)((char*)d_ws + 65536 + 131072 + 1048576);   // 64KB

    hipLaunchKernelGGL(pack_b_kernel, dim3(64), dim3(256), 0, stream, V_w, pB);
    hipLaunchKernelGGL(fkp_kernel, dim3(B_), dim3(D_), 0, stream, h_tm1, Wa_w, Wa_b, fkp, z_num);
    hipLaunchKernelGGL(main_kernel, dim3(B_ * (T_ / TILE_T)), dim3(256), 0, stream,
                       H_r, mask, fkp, pB, V_b, vvec, c, beta_raw, z_num);
    hipLaunchKernelGGL(finalize_kernel, dim3(B_), dim3(256), 0, stream, beta_raw, mask, z_num, out);
}
